// AttentionTracker_89197880803352
// MI455X (gfx1250) — matrix operand order repaired and checked by reading, not compiled, not run on hardware
//
#include <hip/hip_runtime.h>

typedef __attribute__((ext_vector_type(16))) _Float16 v16h;
typedef __attribute__((ext_vector_type(8)))  _Float16 v8h;
typedef __attribute__((ext_vector_type(8)))  float    v8f;

// Monotonic float->uint key: preserves total order of floats as unsigned ints.
__device__ __forceinline__ unsigned fkey(float f) {
  unsigned u = __float_as_uint(f);
  return u ^ ((unsigned)((int)u >> 31) | 0x80000000u);
}

// ---------------------------------------------------------------------------
// Zero the per-row (maxsim, argmax) aggregation array (key 0 == very small).
__global__ void init_agg_kernel(unsigned long long* __restrict__ agg, int n) {
  int i = blockIdx.x * blockDim.x + threadIdx.x;
  if (i < n) agg[i] = 0ull;
}

// ---------------------------------------------------------------------------
// MLP embed + L2 normalize; emit f16 embeddings (row-major N x 64).
// 256 threads = 4 rows x 64 lanes-per-row.
__global__ __launch_bounds__(256) void embed_kernel(
    const float* __restrict__ x, int N,
    const float* __restrict__ W1, const float* __restrict__ b1,
    const float* __restrict__ W2, const float* __restrict__ b2,
    _Float16* __restrict__ out) {
  __shared__ float sW1[4 * 64];
  __shared__ float sB1[64];
  __shared__ float sW2[64 * 64];
  __shared__ float sB2[64];
  __shared__ float sH[4][64];
  __shared__ float sRed[4][64];

  int tid = threadIdx.x;
  for (int i = tid; i < 64 * 64; i += 256) sW2[i] = W2[i];
  sW1[tid] = W1[tid];                       // 256 == 4*64
  if (tid < 64) { sB1[tid] = b1[tid]; sB2[tid] = b2[tid]; }
  __syncthreads();

  int g = tid >> 6, t = tid & 63;
  int row = blockIdx.x * 4 + g;             // grid sized so row < N always

  float x0 = x[row * 4 + 0], x1 = x[row * 4 + 1];
  float x2 = x[row * 4 + 2], x3 = x[row * 4 + 3];
  float h = x0 * sW1[t] + x1 * sW1[64 + t] + x2 * sW1[128 + t] +
            x3 * sW1[192 + t] + sB1[t];
  h = fmaxf(h, 0.0f);
  sH[g][t] = h;
  __syncthreads();

  float e = sB2[t];
#pragma unroll 8
  for (int k = 0; k < 64; ++k) e += sH[g][k] * sW2[k * 64 + t];

  sRed[g][t] = e * e;
  __syncthreads();
  for (int off = 32; off > 0; off >>= 1) {
    if (t < off) sRed[g][t] += sRed[g][t + off];
    __syncthreads();
  }
  float inv = 1.0f / fmaxf(sqrtf(sRed[g][0]), 1e-12f);
  out[row * 64 + t] = (_Float16)(e * inv);
}

// ---------------------------------------------------------------------------
// sim = et @ et1^T via v_wmma_f32_16x16x32_f16 (K=64 -> two chained WMMAs).
// One wave computes a 16(M) x 64(N) strip (A regs reused over 4 B tiles).
// Fused per-row max/argmax via packed u64 atomicMax.
__global__ __launch_bounds__(256) void sim_kernel(
    const _Float16* __restrict__ et, const _Float16* __restrict__ et1,
    float* __restrict__ sim, unsigned long long* __restrict__ agg,
    int Nt1) {
  int lane = threadIdx.x & 31;
  int wave = threadIdx.x >> 5;
  int m0 = blockIdx.x * 16;
  int n0 = (blockIdx.y * 8 + wave) * 64;
  int r  = lane & 15;     // M row (A) / N col (B,C)
  int hi = lane >> 4;     // lane half

  // A 16x32 f16 as loaded here (same K order as B): lane half hi holds contiguous K = 16hi..16hi+15
  const _Float16* arow = et + (size_t)(m0 + r) * 64;
  v16h a0 = *(const v16h*)(arow +      16 * hi);
  v16h a1 = *(const v16h*)(arow + 32 + 16 * hi);

  v8f c[4];
#pragma unroll
  for (int j = 0; j < 4; ++j) {
    // B 32x16 as loaded here (same K order as A): lane half hi holds contiguous K = 16hi..16hi+15
    const _Float16* brow = et1 + (size_t)(n0 + j * 16 + r) * 64 + 16 * hi;
    v16h b0 = *(const v16h*)(brow);
    v16h b1 = *(const v16h*)(brow + 32);
    v8f z = {};
    c[j] = __builtin_amdgcn_wmma_f32_16x16x32_f16(false, a0, false, b0,
                                                  (short)0, z, false, false);
    c[j] = __builtin_amdgcn_wmma_f32_16x16x32_f16(false, a1, false, b1,
                                                  (short)0, c[j], false, false);
  }

  // C/D layout: VGPR i, lane l -> row = i + 8*hi, col = (l&15).
#pragma unroll
  for (int i = 0; i < 8; ++i) {
    int row = m0 + i + hi * 8;
    unsigned long long best = 0ull;
#pragma unroll
    for (int j = 0; j < 4; ++j) {
      float v  = c[j][i];
      int  col = n0 + j * 16 + r;
      __builtin_nontemporal_store(v, sim + (size_t)row * Nt1 + col);
      unsigned long long p =
          ((unsigned long long)fkey(v) << 32) | (unsigned)(~col);
      best = best > p ? best : p;
    }
    // reduce max over the 16 column lanes of this half (xor 1,2,4,8 stays in-half)
#pragma unroll
    for (int m = 8; m >= 1; m >>= 1) {
      unsigned long long o = __shfl_xor(best, m, 32);
      best = best > o ? best : o;
    }
    if (r == 0) atomicMax(&agg[row], best);
  }
}

// ---------------------------------------------------------------------------
// Bitonic sort rows by (maxsim desc, row asc), then serial greedy matching.
// One workgroup; keys (u64) + used bytes in dynamic LDS.
__global__ __launch_bounds__(1024) void match_kernel(
    const unsigned long long* __restrict__ agg,
    int* __restrict__ matches, int N, unsigned thrKey) {
  extern __shared__ unsigned long long smem[];
  unsigned long long* keys = smem;                       // N entries
  unsigned char* used = (unsigned char*)(smem + N);      // N bytes
  int tid = threadIdx.x;

  for (int i = tid; i < N; i += 1024) {
    unsigned long long pk = agg[i];
    unsigned long long key = pk >> 32;
    unsigned long long col = (~(unsigned)pk) & 0x1FFFull;
    // tie-break: ~row so equal keys sort smaller-row-first in descending order
    keys[i] = (key << 32) |
              (((unsigned long long)((~i) & 0x1FFF)) << 13) | col;
    used[i] = 0;
    matches[i] = -1;
  }
  __syncthreads();

  for (int k = 2; k <= N; k <<= 1) {
    for (int j = k >> 1; j > 0; j >>= 1) {
      for (int i = tid; i < N; i += 1024) {
        int l = i ^ j;
        if (l > i) {
          unsigned long long a = keys[i], b = keys[l];
          bool swp = ((i & k) == 0) ? (a < b) : (a > b);  // descending
          if (swp) { keys[i] = b; keys[l] = a; }
        }
      }
      __syncthreads();
    }
  }

  if (tid == 0) {
    for (int i = 0; i < N; ++i) {
      unsigned long long s = keys[i];
      unsigned key = (unsigned)(s >> 32);
      int col = (int)(s & 0x1FFF);
      int row = (int)((~(unsigned)(s >> 13)) & 0x1FFF);
      if (key > thrKey && !used[col]) {
        matches[row] = col;
        used[col] = 1;
      }
    }
  }
}

// ---------------------------------------------------------------------------
extern "C" void kernel_launch(void* const* d_in, const int* in_sizes, int n_in,
                              void* d_out, int out_size, void* d_ws, size_t ws_size,
                              hipStream_t stream) {
  const float* det_t  = (const float*)d_in[0];
  const float* det_t1 = (const float*)d_in[1];
  const float* W1 = (const float*)d_in[2];
  const float* b1 = (const float*)d_in[3];
  const float* W2 = (const float*)d_in[4];
  const float* b2 = (const float*)d_in[5];
  int Nt  = in_sizes[0] / 4;   // 8192
  int Nt1 = in_sizes[1] / 4;   // 8192

  char* ws = (char*)d_ws;
  _Float16* et  = (_Float16*)ws;                                   // Nt*64 f16
  _Float16* et1 = (_Float16*)(ws + (size_t)Nt * 64 * 2);           // Nt1*64 f16
  unsigned long long* agg =
      (unsigned long long*)(ws + (size_t)(Nt + Nt1) * 64 * 2);     // Nt u64

  int*   matches = (int*)d_out;            // first Nt ints (bit-exact int32)
  float* sim     = (float*)d_out + Nt;     // Nt x Nt1 fp32

  init_agg_kernel<<<(Nt + 255) / 256, 256, 0, stream>>>(agg, Nt);
  embed_kernel<<<Nt  / 4, 256, 0, stream>>>(det_t,  Nt,  W1, b1, W2, b2, et);
  embed_kernel<<<Nt1 / 4, 256, 0, stream>>>(det_t1, Nt1, W1, b1, W2, b2, et1);

  dim3 grid(Nt / 16, Nt1 / 512);           // 8 waves/block, 64 cols/wave
  sim_kernel<<<grid, 256, 0, stream>>>(et, et1, sim, agg, Nt1);

  union { float f; unsigned u; } cv; cv.f = 0.3f;
  unsigned thrKey = cv.u ^ ((unsigned)((int)cv.u >> 31) | 0x80000000u);
  size_t shmem = (size_t)Nt * sizeof(unsigned long long) + (size_t)Nt;
  match_kernel<<<1, 1024, shmem, stream>>>(agg, matches, Nt, thrKey);
}
